// Regressor_38749194944899
// MI455X (gfx1250) — matrix operand order repaired and checked by reading, not compiled, not run on hardware
//
#include <hip/hip_runtime.h>

// ---------------------------------------------------------------------------
// Problem constants (from the reference)
// ---------------------------------------------------------------------------
constexpr int cH   = 224;
constexpr int cW   = 608;
constexpr int cHW  = cH * cW;          // 136192 pixels
constexpr int CLS0 = 16, CLS1 = 12, CLS2 = 10;
constexpr int PAD1 = 2,  PAD2 = 3;
constexpr int C12  = CLS0 * CLS1;      // 192
constexpr int C123 = C12 * CLS2;       // 1920
constexpr int SUPER = 384;
constexpr int CFACT = C123 / SUPER;    // 5

typedef __attribute__((ext_vector_type(16))) _Float16 v16h;
typedef __attribute__((ext_vector_type(8)))  float    v8f;

__device__ __forceinline__ float lrelu(float v) { return v > 0.0f ? v : 0.01f * v; }

// ---------------------------------------------------------------------------
// Kernel 1: stage-1 dense per-line MLP with WMMA (f16 in, f32 accum).
// One wave per (line h, 16-pixel tile). K=32 == one v_wmma_f32_16x16x32_f16
// per 16-wide output tile. Layers: 32->32 (lrelu) ->32 (lrelu) ->16 ->argmax.
//
// A layout (16-bit A 16x32, per ISA 7.12.2): lane L holds row M=L&15;
//   elements 0..7  = K = (L<16 ? 0..7 : 8..15)
//   elements 8..15 = K = (L<16 ? 16..23 : 24..31)
// B layout (16-bit B 32x16): lane L holds column N=L&15;
//   elements 0..7 = K = (L<16 ? 0..7 : 8..15), 8..15 = K = (L<16 ? 16..23 : 24..31)
// D layout (f32 16x16): lane L, vgpr r -> (M = r + (L<16?0:8), N = L&15)
// ---------------------------------------------------------------------------
__global__ __launch_bounds__(32)
void stage1_wmma_kernel(const float* __restrict__ x_in,
                        const float* __restrict__ w1_0, const float* __restrict__ b1_0,
                        const float* __restrict__ w1_1, const float* __restrict__ b1_1,
                        const float* __restrict__ w1_2, const float* __restrict__ b1_2,
                        int* __restrict__ inds1)
{
    __shared__ float smem[16][32];     // [pixel-in-tile][channel]

    const int lane = threadIdx.x;      // 0..31, one full wave
    const int h    = blockIdx.y;
    const int w0   = blockIdx.x * 16;
    const int m    = lane & 15;                    // A row / B column
    const int kb   = (lane < 16) ? 0 : 8;          // A K base
    const int krow = (lane < 16) ? 0 : 8;          // B K base

    // ---- load A: channels 0..31 of x_in at 16 consecutive pixels
    v16h A;
    const float* xp = x_in + (long)h * cW + w0 + m;
    #pragma unroll
    for (int e = 0; e < 8; ++e) {
        A[e]     = (_Float16)xp[(long)(kb + e)      * cHW];
        A[e + 8] = (_Float16)xp[(long)(kb + e + 16) * cHW];
    }

    // ---- layers 0 and 1: 32 -> 32, leaky relu
    #pragma unroll
    for (int layer = 0; layer < 2; ++layer) {
        const float* wl = (layer == 0) ? w1_0 : w1_1;   // (H, co=32, ci=32), i contiguous
        const float* bl = (layer == 0) ? b1_0 : b1_1;   // (H, 32)

        v8f d0 = {}; v8f d1 = {};
        {   // N tile 0: outputs o = 0..15    B[K=i][N=o] = wl[(h*32 + o)*32 + i]
            const float* bp = wl + ((long)h * 32 + m) * 32 + krow;
            v16h B;
            #pragma unroll
            for (int e = 0; e < 16; ++e) B[e] = (_Float16)bp[(e < 8) ? e : e + 8];
            d0 = __builtin_amdgcn_wmma_f32_16x16x32_f16(false, A, false, B,
                                                        (short)0, d0, false, false);
        }
        {   // N tile 1: outputs o = 16..31
            const float* bp = wl + ((long)h * 32 + 16 + m) * 32 + krow;
            v16h B;
            #pragma unroll
            for (int e = 0; e < 16; ++e) B[e] = (_Float16)bp[(e < 8) ? e : e + 8];
            d1 = __builtin_amdgcn_wmma_f32_16x16x32_f16(false, A, false, B,
                                                        (short)0, d1, false, false);
        }

        // ---- relayout D(f32) -> A(f16) via LDS, fusing bias + lrelu
        const int mb = (lane < 16) ? 0 : 8;
        __syncthreads();
        #pragma unroll
        for (int r = 0; r < 8; ++r) {
            smem[mb + r][m]      = d0[r];
            smem[mb + r][16 + m] = d1[r];
        }
        __syncthreads();
        const float* bias = bl + (long)h * 32;
        #pragma unroll
        for (int e = 0; e < 8; ++e) {
            const int c0 = kb + e, c1 = kb + e + 16;
            A[e]     = (_Float16)lrelu(smem[m][c0] + bias[c0]);
            A[e + 8] = (_Float16)lrelu(smem[m][c1] + bias[c1]);
        }
    }

    // ---- layer 2: 32 -> 16 logits, then per-pixel argmax
    v8f d2 = {};
    {
        const float* bp = w1_2 + ((long)h * 16 + m) * 32 + krow;  // (H, 16, 32)
        v16h B;
        #pragma unroll
        for (int e = 0; e < 16; ++e) B[e] = (_Float16)bp[(e < 8) ? e : e + 8];
        d2 = __builtin_amdgcn_wmma_f32_16x16x32_f16(false, A, false, B,
                                                    (short)0, d2, false, false);
    }
    const int mb = (lane < 16) ? 0 : 8;
    __syncthreads();
    #pragma unroll
    for (int r = 0; r < 8; ++r) smem[mb + r][m] = d2[r];
    __syncthreads();

    if (lane < 16) {
        const float* bias = b1_2 + (long)h * 16;
        float best = smem[lane][0] + bias[0];
        int   bi   = 0;
        #pragma unroll
        for (int o = 1; o < 16; ++o) {
            const float v = smem[lane][o] + bias[o];
            if (v > best) { best = v; bi = o; }    // first-max tie-break (jnp.argmax)
        }
        inds1[(long)h * cW + w0 + lane] = bi;
    }
}

// ---------------------------------------------------------------------------
// CondMul primitives: one wave == one pixel. The gathered 32x32 weight matrix
// (w[n][i][o], o contiguous) is streamed as 8 float4 per lane -> the wave
// reads contiguous 512B bursts (fully coalesced global_load_b128). Activation
// broadcast via __shfl, partial sums reduced with __shfl_xor. No LDS.
// ---------------------------------------------------------------------------
__device__ __forceinline__ float condmul_32x32(const float* __restrict__ w,
                                               const float* __restrict__ b,
                                               long n, float x, int lane)
{
    const float4* wp = (const float4*)(w + n * 1024);
    float4 acc = make_float4(0.f, 0.f, 0.f, 0.f);
    const int col4 = lane & 7;      // which float4 of a row (out columns 4*col4..+3)
    const int irow = lane >> 3;     // row-group 0..3
    #pragma unroll
    for (int p = 0; p < 8; ++p) {
        const int    i  = irow + (p << 2);         // input channel
        const float4 wv = wp[i * 8 + col4];
        const float  xi = __shfl(x, i);
        acc.x = fmaf(xi, wv.x, acc.x);
        acc.y = fmaf(xi, wv.y, acc.y);
        acc.z = fmaf(xi, wv.z, acc.z);
        acc.w = fmaf(xi, wv.w, acc.w);
    }
    #pragma unroll
    for (int off = 8; off <= 16; off <<= 1) {      // merge the 4 row-groups
        acc.x += __shfl_xor(acc.x, off);
        acc.y += __shfl_xor(acc.y, off);
        acc.z += __shfl_xor(acc.z, off);
        acc.w += __shfl_xor(acc.w, off);
    }
    // redistribute: element j lives in lane j>>2, component j&3
    const int   src = lane >> 2;
    const float v0 = __shfl(acc.x, src);
    const float v1 = __shfl(acc.y, src);
    const float v2 = __shfl(acc.z, src);
    const float v3 = __shfl(acc.w, src);
    const int c = lane & 3;
    const float r = (c == 0) ? v0 : (c == 1) ? v1 : (c == 2) ? v2 : v3;
    return r + b[n * 32 + lane];
}

// 32 -> 16 variant; every lane ends up holding logit[lane & 15]
__device__ __forceinline__ float condmul_32x16(const float* __restrict__ w,
                                               const float* __restrict__ b,
                                               long n, float x, int lane)
{
    const float4* wp = (const float4*)(w + n * 512);
    float4 acc = make_float4(0.f, 0.f, 0.f, 0.f);
    const int col4 = lane & 3;      // 4 float4 per 16-wide row
    const int irow = lane >> 2;     // row-group 0..7
    #pragma unroll
    for (int p = 0; p < 4; ++p) {
        const int    i  = irow + (p << 3);
        const float4 wv = wp[i * 4 + col4];
        const float  xi = __shfl(x, i);
        acc.x = fmaf(xi, wv.x, acc.x);
        acc.y = fmaf(xi, wv.y, acc.y);
        acc.z = fmaf(xi, wv.z, acc.z);
        acc.w = fmaf(xi, wv.w, acc.w);
    }
    #pragma unroll
    for (int off = 4; off <= 16; off <<= 1) {
        acc.x += __shfl_xor(acc.x, off);
        acc.y += __shfl_xor(acc.y, off);
        acc.z += __shfl_xor(acc.z, off);
        acc.w += __shfl_xor(acc.w, off);
    }
    const int   src = lane >> 2;    // lanes 4..7 replicate groups 0..3
    const float v0 = __shfl(acc.x, src);
    const float v1 = __shfl(acc.y, src);
    const float v2 = __shfl(acc.z, src);
    const float v3 = __shfl(acc.w, src);
    const int c = lane & 3;
    const float r = (c == 0) ? v0 : (c == 1) ? v1 : (c == 2) ? v2 : v3;
    return r + b[n * 16 + (lane & 15)];
}

// argmax over 16 logits (value held as logit[lane&15]); first-index tie-break
__device__ __forceinline__ int argmax16(float val, int lane)
{
    int idx = lane & 15;
    #pragma unroll
    for (int off = 1; off < 16; off <<= 1) {
        const float ov = __shfl_xor(val, off);
        const int   oi = __shfl_xor(idx, off);
        if (ov > val || (ov == val && oi < idx)) { val = ov; idx = oi; }
    }
    return idx;
}

__device__ __forceinline__ float wave_sum32(float v)
{
    #pragma unroll
    for (int off = 1; off < 32; off <<= 1) v += __shfl_xor(v, off);
    return v;
}

__device__ __forceinline__ int iclamp(int v, int lo, int hi)
{
    return v < lo ? lo : (v > hi ? hi : v);
}

// ---------------------------------------------------------------------------
// Kernel 2: fused CondMul stages 2, 3 and the regressor. One wave per pixel.
// This kernel is HBM/L2 bound on the gathered weight matrices (~25 KB/pixel),
// so everything is built around coalesced b128 weight streaming.
// ---------------------------------------------------------------------------
__global__ __launch_bounds__(256)
void condmul_stages_kernel(const float* __restrict__ x_in,
                           const int*   __restrict__ inds1_buf,
                           const float* __restrict__ w2_0, const float* __restrict__ b2_0,
                           const float* __restrict__ w2_1, const float* __restrict__ b2_1,
                           const float* __restrict__ w2_2, const float* __restrict__ b2_2,
                           const float* __restrict__ w3_0, const float* __restrict__ b3_0,
                           const float* __restrict__ w3_1, const float* __restrict__ b3_1,
                           const float* __restrict__ w3_2, const float* __restrict__ b3_2,
                           const float* __restrict__ wr0,  const float* __restrict__ br0,
                           const float* __restrict__ wr1,  const float* __restrict__ br1,
                           float* __restrict__ out)
{
    const int wave = (int)((blockIdx.x * blockDim.x + threadIdx.x) >> 5);
    const int lane = threadIdx.x & 31;
    if (wave >= cHW) return;

    const int pix = wave;
    const int h   = pix / cW;

    // ---- stage 2: CondMul chain conditioned on (line, class1)
    const int  i1 = inds1_buf[pix];
    const long n2 = (long)h * CLS0 + i1;
    float x = x_in[(long)(32 + lane) * cHW + pix];
    x = lrelu(condmul_32x32(w2_0, b2_0, n2, x, lane));
    x = lrelu(condmul_32x32(w2_1, b2_1, n2, x, lane));
    const int inds2     = argmax16(condmul_32x16(w2_2, b2_2, n2, x, lane), lane);
    const int inds12    = i1 * CLS1 + (inds2 - PAD1);          // unclipped (ref keeps raw)
    const long n3       = (long)h * C12 + iclamp(inds12, 0, C12 - 1);

    // ---- stage 3: CondMul chain conditioned on (line, class12)
    x = x_in[(long)(64 + lane) * cHW + pix];
    x = lrelu(condmul_32x32(w3_0, b3_0, n3, x, lane));
    x = lrelu(condmul_32x32(w3_1, b3_1, n3, x, lane));
    const int inds3   = argmax16(condmul_32x16(w3_2, b3_2, n3, x, lane), lane);
    const int inds123 = iclamp(inds12 * CLS2 + (inds3 - PAD2), 0, C123 - 1);

    // ---- regressor at the predicted leaf
    const long nsup  = (long)h * SUPER + inds123 / CFACT;
    const long nleaf = (long)h * C123 + inds123;
    x = x_in[(long)(96 + lane) * cHW + pix];
    x = lrelu(condmul_32x32(wr0, br0, nsup, x, lane));
    const float r = wave_sum32(x * wr1[nleaf * 32 + lane]) + br1[nleaf];

    if (lane == 0) {
        const float o = ((float)inds123 + r) * (1.0f / (float)C123);
        out[pix] = (o - 0.1f) * (1.0f / 0.8f);
    }
}

// ---------------------------------------------------------------------------
// Entry point
// ---------------------------------------------------------------------------
extern "C" void kernel_launch(void* const* d_in, const int* in_sizes, int n_in,
                              void* d_out, int out_size, void* d_ws, size_t ws_size,
                              hipStream_t stream)
{
    (void)in_sizes; (void)n_in; (void)out_size; (void)ws_size;

    const float* x_in = (const float*)d_in[0];
    const float* w1_0 = (const float*)d_in[1];
    const float* b1_0 = (const float*)d_in[2];
    const float* w1_1 = (const float*)d_in[3];
    const float* b1_1 = (const float*)d_in[4];
    const float* w1_2 = (const float*)d_in[5];
    const float* b1_2 = (const float*)d_in[6];
    const float* w2_0 = (const float*)d_in[7];
    const float* b2_0 = (const float*)d_in[8];
    const float* w2_1 = (const float*)d_in[9];
    const float* b2_1 = (const float*)d_in[10];
    const float* w2_2 = (const float*)d_in[11];
    const float* b2_2 = (const float*)d_in[12];
    const float* w3_0 = (const float*)d_in[13];
    const float* b3_0 = (const float*)d_in[14];
    const float* w3_1 = (const float*)d_in[15];
    const float* b3_1 = (const float*)d_in[16];
    const float* w3_2 = (const float*)d_in[17];
    const float* b3_2 = (const float*)d_in[18];
    const float* wr0  = (const float*)d_in[19];
    const float* br0  = (const float*)d_in[20];
    const float* wr1  = (const float*)d_in[21];
    const float* br1  = (const float*)d_in[22];
    float* out = (float*)d_out;

    int* inds1 = (int*)d_ws;   // cHW ints of scratch (~545 KB)

    // stage 1: WMMA per-line MLP -> inds1
    dim3 g1(cW / 16, cH);
    stage1_wmma_kernel<<<g1, 32, 0, stream>>>(x_in, w1_0, b1_0, w1_1, b1_1,
                                              w1_2, b1_2, inds1);

    // stages 2/3 + regressor: one wave per pixel, 8 waves per 256-thread block
    const int nblocks = (cHW + 7) / 8;
    condmul_stages_kernel<<<nblocks, 256, 0, stream>>>(
        x_in, inds1,
        w2_0, b2_0, w2_1, b2_1, w2_2, b2_2,
        w3_0, b3_0, w3_1, b3_1, w3_2, b3_2,
        wr0, br0, wr1, br1, out);
}
